// ResGCN_2645699854451
// MI455X (gfx1250) — compile-verified, matrix operand order checked by reading, not run on hardware
//
#include <hip/hip_runtime.h>
#include <hip/hip_bf16.h>
#include <math.h>

typedef __attribute__((ext_vector_type(16))) _Float16 v16h;
typedef __attribute__((ext_vector_type(8)))  _Float16 v8h;
typedef __attribute__((ext_vector_type(4)))  _Float16 v4h;
typedef __attribute__((ext_vector_type(8)))  float    v8f;

#define NN_NODES 50000
#define NN_EDGES 640000
#define F_IN     256
#define HID      128
#define N_GRAPHS 128
#define BN_EPS   1e-5f

// ---------------------------------------------------------------------------
// Utility kernels
// ---------------------------------------------------------------------------
__global__ void zero_kernel(float* __restrict__ p, int n) {
    int i = blockIdx.x * 256 + threadIdx.x;
    if (i < n) p[i] = 0.0f;
}

// One-time weight prep: W[K][128] f32 (row-major) -> Bt[128][K] f16 (transposed)
__global__ void transpose_f16_kernel(const float* __restrict__ W,
                                     _Float16* __restrict__ Bt, int K) {
    int idx = blockIdx.x * 256 + threadIdx.x;      // coalesced read of W
    if (idx < K * 128) {
        int k = idx >> 7, n = idx & 127;
        Bt[(size_t)n * K + k] = (_Float16)W[idx];
    }
}

// Column-wise sum / sumsq over rows. blockDim.x == F (128 or 256).
// RELU folds the previous layer's activation on read.
template <bool RELU>
__global__ void bn_stats_kernel(const float* __restrict__ h,
                                float* __restrict__ sum, float* __restrict__ sumsq,
                                int rows, int F) {
    int f = threadIdx.x;
    float s = 0.0f, s2 = 0.0f;
    for (int r = blockIdx.x; r < rows; r += gridDim.x) {
        float v = h[(size_t)r * F + f];
        if (RELU) v = fmaxf(v, 0.0f);
        s  += v;
        s2 += v * v;
    }
    atomicAdd(&sum[f], s);
    atomicAdd(&sumsq[f], s2);
}

// scale = g*rstd ; shift = b - mu*g*rstd   (so norm(x) = x*scale + shift)
__global__ void bn_finalize_kernel(const float* __restrict__ sum,
                                   const float* __restrict__ sumsq,
                                   const float* __restrict__ g,
                                   const float* __restrict__ b,
                                   float* __restrict__ scale,
                                   float* __restrict__ shift,
                                   float invN) {
    int f = threadIdx.x;
    float mu   = sum[f] * invN;
    float var  = sumsq[f] * invN - mu * mu;
    float rstd = rsqrtf(var + BN_EPS);
    float sc   = g[f] * rstd;
    scale[f] = sc;
    shift[f] = b[f] - mu * sc;
}

__global__ void deg_count_kernel(const int* __restrict__ dst, float* __restrict__ deg, int E) {
    int e = blockIdx.x * 256 + threadIdx.x;
    if (e < E) atomicAdd(&deg[dst[e]], 1.0f);
}

__global__ void deg_finalize_kernel(const float* __restrict__ deg,
                                    float* __restrict__ d_is,
                                    float* __restrict__ self_norm, int N) {
    int i = blockIdx.x * 256 + threadIdx.x;
    if (i < N) {
        float d = deg[i] + 1.0f;
        d_is[i]      = rsqrtf(d);
        self_norm[i] = 1.0f / d;
    }
}

// ---------------------------------------------------------------------------
// WMMA GEMM with fused BN on A and fused self-loop epilogue:
//   hW  = (act(A)*scale+shift) @ Bt^T           (always written)
//   out2 = hW*self_norm[m] + bias[n]            (if SELF; aggregation init)
// A staged in LDS (f16); B fragments read directly from pre-transposed f16
// weights in global (L2/WGP$-resident, 2x global_load_b128 per WMMA).
// grid.x = M/16 ; block = 256 = 8 waves ; wave w -> output cols 16w..16w+15.
// Fragment layouts per CDNA5 ISA 7.12.2.
// ---------------------------------------------------------------------------
template <int K, bool RELU, bool SELF>
__global__ __launch_bounds__(256)
void gemm_bn_wmma(const float* __restrict__ A,
                  const float* __restrict__ scale,
                  const float* __restrict__ shift,
                  const _Float16* __restrict__ Bt,   // [128][K] f16
                  float* __restrict__ out,           // hW
                  const float* __restrict__ self_norm,
                  const float* __restrict__ bias,
                  float* __restrict__ out2) {
    __shared__ __align__(16) _Float16 sA[16 * K];

    const int tid   = threadIdx.x;
    const int wave  = tid >> 5;
    const int lane  = tid & 31;
    const int col   = lane & 15;            // A row sub-index AND output col sub-index
    const int hsel  = (lane >= 16) ? 8 : 0; // K-offset of frags; M-offset of C/D
    const int n     = wave * 16 + col;      // output column
    const int tileM = blockIdx.x * 16;

    // Stage + BN-transform the 16xK A tile into LDS (f16): float4 reads, v4h stores
    {
        constexpr int CH = K / 16;          // per-thread contiguous chunk (16 or 8)
        const int r  = tid >> 4;
        const int k0 = (tid & 15) * CH;
        const float* __restrict__ srcp = A + (size_t)(tileM + r) * K + k0;
        _Float16* __restrict__ dstl = sA + r * K + k0;
#pragma unroll
        for (int k = 0; k < CH; k += 4) {
            float4 v = *(const float4*)(srcp + k);
            if (RELU) {
                v.x = fmaxf(v.x, 0.0f); v.y = fmaxf(v.y, 0.0f);
                v.z = fmaxf(v.z, 0.0f); v.w = fmaxf(v.w, 0.0f);
            }
            v4h h;
            h[0] = (_Float16)(v.x * scale[k0 + k + 0] + shift[k0 + k + 0]);
            h[1] = (_Float16)(v.y * scale[k0 + k + 1] + shift[k0 + k + 1]);
            h[2] = (_Float16)(v.z * scale[k0 + k + 2] + shift[k0 + k + 2]);
            h[3] = (_Float16)(v.w * scale[k0 + k + 3] + shift[k0 + k + 3]);
            *(v4h*)(dstl + k) = h;
        }
    }
    __syncthreads();

    v8f c = {};
    const _Float16* __restrict__ brow = Bt + (size_t)n * K;   // column n of B
#pragma unroll
    for (int kc = 0; kc < K; kc += 32) {
        // A fragment: row=col, K offsets {hsel..hsel+7, hsel+16..hsel+23}
        const _Float16* pa = &sA[col * K + kc + hsel];
        v8h a0 = *(const v8h*)(pa);
        v8h a1 = *(const v8h*)(pa + 16);
        v16h av = __builtin_shufflevector(a0, a1, 0,1,2,3,4,5,6,7,8,9,10,11,12,13,14,15);

        // B fragment: same K offsets, contiguous in transposed weights (global)
        const _Float16* pb = brow + kc + hsel;
        v8h b0 = *(const v8h*)(pb);
        v8h b1 = *(const v8h*)(pb + 16);
        v16h bv = __builtin_shufflevector(b0, b1, 0,1,2,3,4,5,6,7,8,9,10,11,12,13,14,15);

        c = __builtin_amdgcn_wmma_f32_16x16x32_f16(false, av, false, bv,
                                                   (short)0, c, false, false);
    }

    // C/D layout: VGPR r -> M = r + (lane>=16 ? 8 : 0), N = lane&15
    const float bn_ = SELF ? bias[n] : 0.0f;
#pragma unroll
    for (int r = 0; r < 8; ++r) {
        int m = tileM + r + hsel;
        float v = c[r];
        out[(size_t)m * 128 + n] = v;
        if (SELF) out2[(size_t)m * 128 + n] = v * self_norm[m] + bn_;
    }
}

// ---------------------------------------------------------------------------
// GCN aggregation: out[dst] += hW[src] * d_is[src]*d_is[dst]
// One wave per edge; each lane handles 4 contiguous features (b128 load).
// ---------------------------------------------------------------------------
__global__ void aggregate_kernel(const float* __restrict__ hW,
                                 const int* __restrict__ src,
                                 const int* __restrict__ dst,
                                 const float* __restrict__ d_is,
                                 float* __restrict__ out, int E) {
    int e = blockIdx.x * 8 + (threadIdx.x >> 5);
    int f = (threadIdx.x & 31) * 4;
    if (e < E) {
        int s = src[e], d = dst[e];
        float w = d_is[s] * d_is[d];
        float4 v = *(const float4*)&hW[(size_t)s * HID + f];
        float* o = &out[(size_t)d * HID + f];
        atomicAdd(o + 0, v.x * w);
        atomicAdd(o + 1, v.y * w);
        atomicAdd(o + 2, v.z * w);
        atomicAdd(o + 3, v.w * w);
    }
}

// global_add_pool with fused ReLU on read (conv output is pre-activation)
__global__ void pool_kernel(const float* __restrict__ h,
                            const int* __restrict__ batch,
                            float* __restrict__ z, int N) {
    int t = blockIdx.x * 256 + threadIdx.x;
    int i = t >> 5, f = (t & 31) * 4;
    if (i < N) {
        int g = batch[i];
        float4 v = *(const float4*)&h[(size_t)i * HID + f];
        float* o = &z[(size_t)g * HID + f];
        atomicAdd(o + 0, fmaxf(v.x, 0.0f));
        atomicAdd(o + 1, fmaxf(v.y, 0.0f));
        atomicAdd(o + 2, fmaxf(v.z, 0.0f));
        atomicAdd(o + 3, fmaxf(v.w, 0.0f));
    }
}

__global__ void bias_relu_kernel(float* __restrict__ z, const float* __restrict__ b) {
    int g = blockIdx.x, f = threadIdx.x;            // grid = 128, block = 128
    float v = z[(size_t)g * HID + f] + b[f];
    z[(size_t)g * HID + f] = fmaxf(v, 0.0f);
}

// logits = BN(z2) @ W_cls + b_cls ; out = sigmoid(logits)
__global__ void final_kernel(const float* __restrict__ z2,
                             const float* __restrict__ scale,
                             const float* __restrict__ shift,
                             const float* __restrict__ Wc,
                             const float* __restrict__ bc,
                             float* __restrict__ out) {
    int g = threadIdx.x;                            // 128 threads, 1 block
    float acc = bc[0];
    for (int h = 0; h < HID; ++h)
        acc += (z2[(size_t)g * HID + h] * scale[h] + shift[h]) * Wc[h];
    out[g] = 1.0f / (1.0f + expf(-acc));
}

// ---------------------------------------------------------------------------
// Host orchestration
// ---------------------------------------------------------------------------
extern "C" void kernel_launch(void* const* d_in, const int* in_sizes, int n_in,
                              void* d_out, int out_size, void* d_ws, size_t ws_size,
                              hipStream_t stream) {
    (void)in_sizes; (void)n_in; (void)out_size; (void)ws_size;

    const float* x       = (const float*)d_in[0];
    const int*   src     = (const int*)d_in[1];
    const int*   dstp    = src + NN_EDGES;
    const int*   batch   = (const int*)d_in[2];
    const float* bn_f_g  = (const float*)d_in[3];
    const float* bn_f_b  = (const float*)d_in[4];
    const float* W_feat  = (const float*)d_in[5];
    const float* b_feat  = (const float*)d_in[6];
    const float* bn_c_g  = (const float*)d_in[7];
    const float* bn_c_b  = (const float*)d_in[8];
    const float* W_conv  = (const float*)d_in[9];
    const float* b_conv  = (const float*)d_in[10];
    const float* bn_fc_g = (const float*)d_in[11];
    const float* bn_fc_b = (const float*)d_in[12];
    const float* W_fc    = (const float*)d_in[13];
    const float* b_fc    = (const float*)d_in[14];
    const float* bn_h_g  = (const float*)d_in[15];
    const float* bn_h_b  = (const float*)d_in[16];
    const float* W_cls   = (const float*)d_in[17];
    const float* b_cls   = (const float*)d_in[18];

    float* ws    = (float*)d_ws;
    float* sum   = ws;                 // 256
    float* sumsq = ws + 256;           // 256
    float* scl   = ws + 512;           // 256
    float* shf   = ws + 768;           // 256
    float* deg   = ws + 1024;          // NN
    float* d_is  = deg  + NN_NODES;    // NN
    float* sfn   = d_is + NN_NODES;    // NN
    float* hW    = sfn  + NN_NODES;                     // NN*128
    float* hA    = hW + (size_t)NN_NODES * HID;         // NN*128
    float* hB    = hA + (size_t)NN_NODES * HID;         // NN*128
    float* z     = hB + (size_t)NN_NODES * HID;         // 128*128
    float* z2    = z  + N_GRAPHS * HID;                 // 128*128

    // f16 transposed-weight region (16B-aligned: float region is 16B multiple)
    _Float16* Bt_feat = (_Float16*)(z2 + N_GRAPHS * HID);   // 128*256
    _Float16* Bt_c0   = Bt_feat + 128 * 256;                // 128*128
    _Float16* Bt_c1   = Bt_c0 + 128 * 128;
    _Float16* Bt_c2   = Bt_c1 + 128 * 128;
    _Float16* Bt_fc   = Bt_c2 + 128 * 128;
    _Float16* Bt_conv[3] = { Bt_c0, Bt_c1, Bt_c2 };

    const int NQ = NN_NODES * 32;      // node-feature grid (4 feats/thread)

    // ---- one-time weight transpose+convert (f32 -> f16, [K,128] -> [128,K]) ----
    transpose_f16_kernel<<<(F_IN * 128 + 255) / 256, 256, 0, stream>>>(W_feat, Bt_feat, F_IN);
    for (int i = 0; i < 3; ++i)
        transpose_f16_kernel<<<(HID * 128 + 255) / 256, 256, 0, stream>>>(
            W_conv + (size_t)i * HID * HID, Bt_conv[i], HID);
    transpose_f16_kernel<<<(HID * 128 + 255) / 256, 256, 0, stream>>>(W_fc, Bt_fc, HID);

    // ---- degree / normalization constants ----
    zero_kernel<<<(NN_NODES + 255) / 256, 256, 0, stream>>>(deg, NN_NODES);
    deg_count_kernel<<<(NN_EDGES + 255) / 256, 256, 0, stream>>>(dstp, deg, NN_EDGES);
    deg_finalize_kernel<<<(NN_NODES + 255) / 256, 256, 0, stream>>>(deg, d_is, sfn, NN_NODES);

    // ---- input BatchNorm (F=256, no relu on x) ----
    zero_kernel<<<2, 256, 0, stream>>>(sum, 512);
    bn_stats_kernel<false><<<256, F_IN, 0, stream>>>(x, sum, sumsq, NN_NODES, F_IN);
    bn_finalize_kernel<<<1, F_IN, 0, stream>>>(sum, sumsq, bn_f_g, bn_f_b, scl, shf,
                                               1.0f / NN_NODES);

    // ---- layer 1: BN(x) @ W_feat (self-loop fused) -> edge aggregate ----
    gemm_bn_wmma<256, false, true><<<NN_NODES / 16, 256, 0, stream>>>(
        x, scl, shf, Bt_feat, hW, sfn, b_feat, hA);
    aggregate_kernel<<<NN_EDGES / 8, 256, 0, stream>>>(hW, src, dstp, d_is, hA, NN_EDGES);

    // ---- 3 conv layers (relu folded into BN-stats read + GEMM A-load) ----
    float* cur = hA;
    float* nxt = hB;
    for (int i = 0; i < 3; ++i) {
        zero_kernel<<<2, 256, 0, stream>>>(sum, 512);
        bn_stats_kernel<true><<<256, HID, 0, stream>>>(cur, sum, sumsq, NN_NODES, HID);
        bn_finalize_kernel<<<1, HID, 0, stream>>>(sum, sumsq,
                                                  bn_c_g + i * HID, bn_c_b + i * HID,
                                                  scl, shf, 1.0f / NN_NODES);
        gemm_bn_wmma<128, true, true><<<NN_NODES / 16, 256, 0, stream>>>(
            cur, scl, shf, Bt_conv[i], hW, sfn, b_conv + i * HID, nxt);
        aggregate_kernel<<<NN_EDGES / 8, 256, 0, stream>>>(hW, src, dstp, d_is, nxt, NN_EDGES);
        float* t = cur; cur = nxt; nxt = t;
    }

    // ---- global add pool (relu on read) ----
    zero_kernel<<<(N_GRAPHS * HID + 255) / 256, 256, 0, stream>>>(z, N_GRAPHS * HID);
    pool_kernel<<<(NQ + 255) / 256, 256, 0, stream>>>(cur, batch, z, NN_NODES);

    // ---- head: BN -> fc (WMMA) -> bias+relu -> BN -> cls -> sigmoid ----
    zero_kernel<<<2, 256, 0, stream>>>(sum, 512);
    bn_stats_kernel<false><<<N_GRAPHS, HID, 0, stream>>>(z, sum, sumsq, N_GRAPHS, HID);
    bn_finalize_kernel<<<1, HID, 0, stream>>>(sum, sumsq, bn_fc_g, bn_fc_b, scl, shf,
                                              1.0f / N_GRAPHS);
    gemm_bn_wmma<128, false, false><<<N_GRAPHS / 16, 256, 0, stream>>>(
        z, scl, shf, Bt_fc, z2, nullptr, nullptr, nullptr);
    bias_relu_kernel<<<N_GRAPHS, HID, 0, stream>>>(z2, b_fc);

    zero_kernel<<<2, 256, 0, stream>>>(sum, 512);
    bn_stats_kernel<false><<<N_GRAPHS, HID, 0, stream>>>(z2, sum, sumsq, N_GRAPHS, HID);
    bn_finalize_kernel<<<1, HID, 0, stream>>>(sum, sumsq, bn_h_g, bn_h_b, scl, shf,
                                              1.0f / N_GRAPHS);
    final_kernel<<<1, N_GRAPHS, 0, stream>>>(z2, scl, shf, W_cls, b_cls, (float*)d_out);
}
